// GRUModel_LN_6055903888161
// MI455X (gfx1250) — matrix operand order repaired and checked by reading, not compiled, not run on hardware
//
#include <hip/hip_runtime.h>
#include <hip/hip_bf16.h>
#include <stdint.h>

// ---------------------------------------------------------------------------
// LayerNorm-GRU (2 layers, shared h), persistent WMMA kernel for gfx1250.
//   B=64, T=512, IN=64, H=512.
//   * 32 persistent WGs x 128 threads; WG g owns H-columns [16g,16g+16) and
//     the matching rows of all three gate blocks of each weight matrix.
//   * Weight slices staged ONCE into LDS via the Tensor Data Mover (TDM,
//     tensor_load_to_lds + s_wait_tensorcnt), falling back to plain copies.
//   * 278 KB LDS/WG (CDNA5: 320 KB/WGP): weights + two bf16 A-matrix buffers.
//   * Phase B computes one shared normalized vector v^ and emits both xn1 and
//     hn1 A-matrices in a single pass (two affine tails) -> one LN pass and
//     two workgroup barriers saved per timestep.
//   * GEMMs: v_wmma_f32_16x16x32_bf16 (bf16 A/B from LDS, f32 acc).
//   * 2 grid barriers per step; ln_ho deferred to consumer as composed LN.
// ---------------------------------------------------------------------------

namespace {

constexpr int Bn  = 64;
constexpr int Tn  = 512;
constexpr int INn = 64;
constexpr int Hn  = 512;
constexpr float EPSf = 1e-5f;

typedef __attribute__((ext_vector_type(16))) __bf16 v16bf;
typedef __attribute__((ext_vector_type(8)))  float  v8f;

union FragBF { uint4 u4[2]; v16bf v; };

__device__ __forceinline__ float wred32(float x) {
#pragma unroll
  for (int off = 16; off > 0; off >>= 1) x += __shfl_xor(x, off, 32);
  return x;
}
__device__ __forceinline__ float sigmf(float x) { return 1.0f / (1.0f + __expf(-x)); }

// out = LN_{g2,b2}( LN_{g1,b1}(src_row) ) -> bf16 LDS [64][512]
__device__ void ln2_to_lds(const float* __restrict__ src,
                           const float* __restrict__ g1, const float* __restrict__ b1,
                           const float* __restrict__ g2, const float* __restrict__ b2,
                           __bf16* __restrict__ lds) {
  const int wid  = threadIdx.x >> 5;
  const int lane = threadIdx.x & 31;
  for (int r = wid * 16; r < wid * 16 + 16; ++r) {
    const float* row = src + r * Hn;
    float u[16];
    float s = 0.f;
#pragma unroll
    for (int i = 0; i < 16; ++i) { u[i] = row[lane * 16 + i]; s += u[i]; }
    s = wred32(s) * (1.f / Hn);
    float s2 = 0.f;
#pragma unroll
    for (int i = 0; i < 16; ++i) { float d = u[i] - s; s2 += d * d; }
    s2 = wred32(s2) * (1.f / Hn);
    const float rs = rsqrtf(s2 + EPSf);
    float vv[16];
    float t = 0.f;
#pragma unroll
    for (int i = 0; i < 16; ++i) {
      int k = lane * 16 + i;
      vv[i] = (u[i] - s) * rs * g1[k] + b1[k];
      t += vv[i];
    }
    t = wred32(t) * (1.f / Hn);
    float t2 = 0.f;
#pragma unroll
    for (int i = 0; i < 16; ++i) { float d = vv[i] - t; t2 += d * d; }
    t2 = wred32(t2) * (1.f / Hn);
    const float rt = rsqrtf(t2 + EPSf);
#pragma unroll
    for (int i = 0; i < 16; ++i) {
      int k = lane * 16 + i;
      lds[r * Hn + k] = (__bf16)((vv[i] - t) * rt * g2[k] + b2[k]);
    }
  }
}

// v = LN_{g1,b1}(src_row); v^ = normalize(v);
// ldsX = v^*gx+bx ; ldsH = v^*gh+bh  (one pass, two affine tails).
__device__ void ln2_dual_to_lds(const float* __restrict__ src,
                                const float* __restrict__ g1, const float* __restrict__ b1,
                                const float* __restrict__ gx, const float* __restrict__ bx,
                                const float* __restrict__ gh, const float* __restrict__ bh,
                                __bf16* __restrict__ ldsX, __bf16* __restrict__ ldsH) {
  const int wid  = threadIdx.x >> 5;
  const int lane = threadIdx.x & 31;
  for (int r = wid * 16; r < wid * 16 + 16; ++r) {
    const float* row = src + r * Hn;
    float u[16];
    float s = 0.f;
#pragma unroll
    for (int i = 0; i < 16; ++i) { u[i] = row[lane * 16 + i]; s += u[i]; }
    s = wred32(s) * (1.f / Hn);
    float s2 = 0.f;
#pragma unroll
    for (int i = 0; i < 16; ++i) { float d = u[i] - s; s2 += d * d; }
    s2 = wred32(s2) * (1.f / Hn);
    const float rs = rsqrtf(s2 + EPSf);
    float vv[16];
    float t = 0.f;
#pragma unroll
    for (int i = 0; i < 16; ++i) {
      int k = lane * 16 + i;
      vv[i] = (u[i] - s) * rs * g1[k] + b1[k];
      t += vv[i];
    }
    t = wred32(t) * (1.f / Hn);
    float t2 = 0.f;
#pragma unroll
    for (int i = 0; i < 16; ++i) { float d = vv[i] - t; t2 += d * d; }
    t2 = wred32(t2) * (1.f / Hn);
    const float rt = rsqrtf(t2 + EPSf);
#pragma unroll
    for (int i = 0; i < 16; ++i) {
      int k = lane * 16 + i;
      const float vh = (vv[i] - t) * rt;
      ldsX[r * Hn + k] = (__bf16)(vh * gx[k] + bx[k]);
      ldsH[r * Hn + k] = (__bf16)(vh * gh[k] + bh[k]);
    }
  }
}

// xn0 = LN_{g,b}(x[:, t, :]) -> bf16 LDS [64][64]
__device__ void lnx_to_lds(const float* __restrict__ x, int t,
                           const float* __restrict__ g, const float* __restrict__ b,
                           __bf16* __restrict__ lds) {
  const int r = threadIdx.x;
  if (r < Bn) {
    const float* row = x + ((size_t)r * Tn + t) * INn;
    float s = 0.f;
    for (int k = 0; k < INn; ++k) s += row[k];
    s *= (1.f / INn);
    float s2 = 0.f;
    for (int k = 0; k < INn; ++k) { float d = row[k] - s; s2 += d * d; }
    s2 *= (1.f / INn);
    const float rs = rsqrtf(s2 + EPSf);
    for (int k = 0; k < INn; ++k)
      lds[r * INn + k] = (__bf16)((row[k] - s) * rs * g[k] + b[k]);
  }
}

// A fragment per ISA 16-bit 16x32 table.
__device__ __forceinline__ v16bf load_fragA(const __bf16* lds, int mbase, int S,
                                            int kk, int lane) {
  FragBF f;
  const int m  = mbase + (lane & 15);
  const int kh = (lane >> 4) * 8;
  f.u4[0] = *reinterpret_cast<const uint4*>(lds + m * S + kk + kh);
  f.u4[1] = *reinterpret_cast<const uint4*>(lds + m * S + kk + 16 + kh);
  return f.v;
}

// B fragment from LDS weight slice [48][KD] (rows = gate*16 + local column).
template <int KD>
__device__ __forceinline__ v16bf load_fragB_lds(const __bf16* ldsW, int G, int lane,
                                                int kk) {
  FragBF f;
  const int rl = G * 16 + (lane & 15);
  const int kb = (lane >> 4) * 8;
  const uint4* p = reinterpret_cast<const uint4*>(ldsW + rl * KD + kk + kb);
  f.u4[0] = p[0];
  f.u4[1] = p[2];
  return f.v;
}

template <int KD>
__device__ __forceinline__ void gemm3(const __bf16* __restrict__ ldsA, int S,
                                      const __bf16* __restrict__ ldsW,
                                      int mbase, int lane, v8f acc[3]) {
#pragma unroll
  for (int kk = 0; kk < KD; kk += 32) {
    v16bf a = load_fragA(ldsA, mbase, S, kk, lane);
#pragma unroll
    for (int G = 0; G < 3; ++G) {
      v16bf bb = load_fragB_lds<KD>(ldsW, G, lane, kk);
      acc[G] = __builtin_amdgcn_wmma_f32_16x16x32_bf16(
          false, a, false, bb, (short)0, acc[G], false, false);
    }
  }
}

__device__ void grid_sync(int* cnt, int* gen, int nb) {
  __threadfence();
  __syncthreads();
  if (threadIdx.x == 0) {
    const int g = __hip_atomic_load(gen, __ATOMIC_RELAXED, __HIP_MEMORY_SCOPE_AGENT);
    const int a = __hip_atomic_fetch_add(cnt, 1, __ATOMIC_ACQ_REL, __HIP_MEMORY_SCOPE_AGENT);
    if (a == nb - 1) {
      __hip_atomic_store(cnt, 0, __ATOMIC_RELAXED, __HIP_MEMORY_SCOPE_AGENT);
      __hip_atomic_fetch_add(gen, 1, __ATOMIC_RELEASE, __HIP_MEMORY_SCOPE_AGENT);
    } else {
      while (__hip_atomic_load(gen, __ATOMIC_ACQUIRE, __HIP_MEMORY_SCOPE_AGENT) == g)
        __builtin_amdgcn_s_sleep(1);
    }
  }
  __syncthreads();
}

// ---------------- Tensor Data Mover staging (CDNA5 TDM) --------------------
#if __has_builtin(__builtin_amdgcn_tensor_load_to_lds)
#define GRU_USE_TDM 1
typedef __attribute__((ext_vector_type(4))) unsigned int v4u;
typedef __attribute__((ext_vector_type(8))) int v8i;
typedef __attribute__((ext_vector_type(4))) int v4i;

// 2D tile: rows x kdim bf16 elements, row stride = kdim (contiguous rows),
// from global gsrc into LDS byte offset lds_off. D# packing per ISA ch.8.
__device__ __forceinline__ void tdm_load_tile(const __bf16* gsrc, unsigned lds_off,
                                              int kdim, int rows) {
  const unsigned long long ga = (unsigned long long)(uintptr_t)gsrc;
  v4u g0;
  g0.x = 1u;                                   // count=1 (valid), user mode
  g0.y = lds_off;                              // lds_addr (bytes)
  g0.z = (unsigned)(ga & 0xffffffffull);       // global_addr[31:0]
  g0.w = (unsigned)(ga >> 32) | (2u << 30);    // global_addr[56:32] | type=2
  v8i g1;
  g1[0] = 0x00010000;                          // data_size = 2 bytes
  g1[1] = (int)((unsigned)kdim << 16);         // tensor_dim0[15:0]
  g1[2] = (int)((unsigned)rows << 16);         // tensor_dim1[15:0]
  g1[3] = (int)((unsigned)kdim << 16);         // tile_dim0
  g1[4] = rows;                                // tile_dim1 (tile_dim2=0)
  g1[5] = kdim;                                // tensor_dim0_stride[31:0]
  g1[6] = 0;
  g1[7] = 0;
  v4i z = {};
#if defined(__clang_major__) && __clang_major__ >= 23
  v8i z8 = {};
  __builtin_amdgcn_tensor_load_to_lds(g0, g1, z, z, z8, 0);
#else
  __builtin_amdgcn_tensor_load_to_lds(g0, g1, z, z, 0);
#endif
}
#endif

// Weight fp32 -> bf16 ([n][k] layout preserved) + barrier re-arm.
__global__ void conv_weights(const float* __restrict__ Wih0, const float* __restrict__ Whh0,
                             const float* __restrict__ Wih1, const float* __restrict__ Whh1,
                             __bf16* o0, __bf16* o1, __bf16* o2, __bf16* o3,
                             int* cnt, int* gen) {
  const size_t i = (size_t)blockIdx.x * blockDim.x + threadIdx.x;
  if (i == 0) { *cnt = 0; *gen = 0; }
  if (i < (size_t)3 * Hn * INn) o0[i] = (__bf16)Wih0[i];
  if (i < (size_t)3 * Hn * Hn) {
    o1[i] = (__bf16)Whh0[i];
    o2[i] = (__bf16)Wih1[i];
    o3[i] = (__bf16)Whh1[i];
  }
}

__global__ __launch_bounds__(128) void gru_persistent(
    const float* __restrict__ x, const int* __restrict__ lengths,
    const float* __restrict__ bih0, const float* __restrict__ bhh0,
    const float* __restrict__ g_ih0, const float* __restrict__ b_ih0,
    const float* __restrict__ g_hh0, const float* __restrict__ b_hh0,
    const float* __restrict__ g_ho0, const float* __restrict__ b_ho0,
    const float* __restrict__ bih1, const float* __restrict__ bhh1,
    const float* __restrict__ g_ih1, const float* __restrict__ b_ih1,
    const float* __restrict__ g_hh1, const float* __restrict__ b_hh1,
    const float* __restrict__ g_ho1, const float* __restrict__ b_ho1,
    const __bf16* __restrict__ Wih0b, const __bf16* __restrict__ Whh0b,
    const __bf16* __restrict__ Wih1b, const __bf16* __restrict__ Whh1b,
    float* __restrict__ u_raw,   // [64][512] layer0 raw (pre ln_ho0) hidden
    float* __restrict__ w_raw,   // [64][512] layer1 raw (pre ln_ho1) hidden
    float* __restrict__ last_raw,// [64][512] captured at t == len-1
    int* cnt, int* gen) {
  // 284,672 bytes of LDS per workgroup (one WG per WGP on CDNA5).
  __shared__ __align__(16) __bf16 ldsA [Bn * Hn];       // 64 KB A-matrix (xn)
  __shared__ __align__(16) __bf16 ldsA2[Bn * Hn];       // 64 KB A-matrix (hn)
  __shared__ __align__(16) __bf16 ldsWhh0[48 * Hn];     // 48 KB
  __shared__ __align__(16) __bf16 ldsWih1[48 * Hn];     // 48 KB
  __shared__ __align__(16) __bf16 ldsWhh1[48 * Hn];     // 48 KB
  __shared__ __align__(16) __bf16 ldsWih0[48 * INn];    // 6 KB

  const int lane  = threadIdx.x & 31;
  const int wid   = threadIdx.x >> 5;
  const int j0    = blockIdx.x * 16;   // owned H-columns [j0, j0+16)
  const int mbase = wid * 16;          // owned batch rows [mbase, mbase+16)
  const int jc    = j0 + (lane & 15);
  const int nb    = gridDim.x;

#pragma unroll
  for (int G = 0; G < 3; ++G) {
    __builtin_prefetch(Whh0b + (size_t)(G * Hn + j0) * Hn, 0, 3);
    __builtin_prefetch(Wih1b + (size_t)(G * Hn + j0) * Hn, 0, 3);
    __builtin_prefetch(Whh1b + (size_t)(G * Hn + j0) * Hn, 0, 3);
  }

  // ---- Stage weight slices global -> LDS once. ----
#ifdef GRU_USE_TDM
  if (wid == 0) {
#pragma unroll
    for (int G = 0; G < 3; ++G) {
      const int n0 = G * Hn + j0;  // 16 consecutive weight rows per gate slice
      tdm_load_tile(Whh0b + (size_t)n0 * Hn,
                    (unsigned)(uintptr_t)(ldsWhh0 + G * 16 * Hn), Hn, 16);
      tdm_load_tile(Wih1b + (size_t)n0 * Hn,
                    (unsigned)(uintptr_t)(ldsWih1 + G * 16 * Hn), Hn, 16);
      tdm_load_tile(Whh1b + (size_t)n0 * Hn,
                    (unsigned)(uintptr_t)(ldsWhh1 + G * 16 * Hn), Hn, 16);
      tdm_load_tile(Wih0b + (size_t)n0 * INn,
                    (unsigned)(uintptr_t)(ldsWih0 + G * 16 * INn), INn, 16);
    }
    __builtin_amdgcn_s_wait_tensorcnt(0);
  }
#else
  for (int idx = threadIdx.x; idx < 48 * 64; idx += 128) {
    const int rl = idx >> 6;
    const int q  = idx & 63;
    const int n  = (rl >> 4) * Hn + j0 + (rl & 15);
    *reinterpret_cast<uint4*>(ldsWhh0 + rl * Hn + q * 8) =
        *reinterpret_cast<const uint4*>(Whh0b + (size_t)n * Hn + q * 8);
    *reinterpret_cast<uint4*>(ldsWih1 + rl * Hn + q * 8) =
        *reinterpret_cast<const uint4*>(Wih1b + (size_t)n * Hn + q * 8);
    *reinterpret_cast<uint4*>(ldsWhh1 + rl * Hn + q * 8) =
        *reinterpret_cast<const uint4*>(Whh1b + (size_t)n * Hn + q * 8);
  }
  for (int idx = threadIdx.x; idx < 48 * 8; idx += 128) {
    const int rl = idx >> 3;
    const int q  = idx & 7;
    const int n  = (rl >> 4) * Hn + j0 + (rl & 15);
    *reinterpret_cast<uint4*>(ldsWih0 + rl * INn + q * 8) =
        *reinterpret_cast<const uint4*>(Wih0b + (size_t)n * INn + q * 8);
  }
#endif
  __syncthreads();

  const float br0 = bih0[jc], bz0 = bih0[Hn + jc], bn0 = bih0[2 * Hn + jc];
  const float cr0 = bhh0[jc], cz0 = bhh0[Hn + jc], cn0 = bhh0[2 * Hn + jc];
  const float br1 = bih1[jc], bz1 = bih1[Hn + jc], bn1 = bih1[2 * Hn + jc];
  const float cr1 = bhh1[jc], cz1 = bhh1[Hn + jc], cn1 = bhh1[2 * Hn + jc];

  for (int t = 0; t < Tn; ++t) {
    // ---------------- Layer 0 ----------------
    // A1: xn0 = ln_ih0(x_t), gi0 = xn0 @ Wih0.T
    lnx_to_lds(x, t, g_ih0, b_ih0, ldsA);
    __syncthreads();
    v8f acc_i[3] = {};
    gemm3<INn>(ldsA, INn, ldsWih0, mbase, lane, acc_i);
    __syncthreads();

    // A2: hn0 = ln_hh0( h )   with h = ln_ho1(w_raw)   (t=0: h = 0 -> b_hh0)
    if (t == 0) {
      for (int idx = threadIdx.x; idx < Bn * Hn; idx += 128)
        ldsA[idx] = (__bf16)b_hh0[idx & (Hn - 1)];
    } else {
      ln2_to_lds(w_raw, g_ho1, b_ho1, g_hh0, b_hh0, ldsA);
    }
    __syncthreads();
    v8f acc_h[3] = {};
    gemm3<Hn>(ldsA, Hn, ldsWhh0, mbase, lane, acc_h);

    // Gates (layer 0) -> raw u (pre ln_ho0)
#pragma unroll
    for (int v = 0; v < 8; ++v) {
      const int M = mbase + v + ((lane >> 4) << 3);
      const float ir = acc_i[0][v] + br0, iz = acc_i[1][v] + bz0, in_ = acc_i[2][v] + bn0;
      const float hr = acc_h[0][v] + cr0, hz = acc_h[1][v] + cz0, hn_ = acc_h[2][v] + cn0;
      const float r  = sigmf(ir + hr);
      const float z  = sigmf(iz + hz);
      const float ng = tanhf(in_ + r * hn_);
      const float hnv = (float)ldsA[M * Hn + jc];  // hn0
      u_raw[M * Hn + jc] = (1.f - z) * ng + z * hnv;
    }
    grid_sync(cnt, gen, nb);

    // ---------------- Layer 1 ----------------
    // One pass: v = ln_ho0(u_raw); xn1 = LN_ih1(v) -> ldsA ; hn1 = LN_hh1(v) -> ldsA2
    ln2_dual_to_lds(u_raw, g_ho0, b_ho0, g_ih1, b_ih1, g_hh1, b_hh1, ldsA, ldsA2);
    __syncthreads();
    v8f acc_i1[3] = {};
    gemm3<Hn>(ldsA, Hn, ldsWih1, mbase, lane, acc_i1);
    v8f acc_h1[3] = {};
    gemm3<Hn>(ldsA2, Hn, ldsWhh1, mbase, lane, acc_h1);

    // Gates (layer 1) -> raw w (pre ln_ho1); capture last at t == len-1.
#pragma unroll
    for (int v = 0; v < 8; ++v) {
      const int M = mbase + v + ((lane >> 4) << 3);
      const float ir = acc_i1[0][v] + br1, iz = acc_i1[1][v] + bz1, in_ = acc_i1[2][v] + bn1;
      const float hr = acc_h1[0][v] + cr1, hz = acc_h1[1][v] + cz1, hn_ = acc_h1[2][v] + cn1;
      const float r  = sigmf(ir + hr);
      const float z  = sigmf(iz + hz);
      const float ng = tanhf(in_ + r * hn_);
      const float hnv = (float)ldsA2[M * Hn + jc];  // hn1
      const float wv_ = (1.f - z) * ng + z * hnv;
      w_raw[M * Hn + jc] = wv_;
      if (t == lengths[M] - 1) last_raw[M * Hn + jc] = wv_;
    }
    grid_sync(cnt, gen, nb);
  }
}

// out = sigmoid( ln_ho1(last_raw) @ fcW.T + fcb )
__global__ void fc_final(const float* __restrict__ last_raw,
                         const float* __restrict__ g, const float* __restrict__ bta,
                         const float* __restrict__ fcW, const float* __restrict__ fcb,
                         float* __restrict__ out) {
  const int b = threadIdx.x;
  if (b >= Bn) return;
  const float* row = last_raw + b * Hn;
  float s = 0.f;
  for (int k = 0; k < Hn; ++k) s += row[k];
  s *= (1.f / Hn);
  float s2 = 0.f;
  for (int k = 0; k < Hn; ++k) { float d = row[k] - s; s2 += d * d; }
  s2 *= (1.f / Hn);
  const float rs = rsqrtf(s2 + EPSf);
  float d0 = fcb[0], d1 = fcb[1];
  for (int k = 0; k < Hn; ++k) {
    const float v = (row[k] - s) * rs * g[k] + bta[k];
    d0 += v * fcW[k];
    d1 += v * fcW[Hn + k];
  }
  out[b * 2 + 0] = sigmf(d0);
  out[b * 2 + 1] = sigmf(d1);
}

}  // namespace

extern "C" void kernel_launch(void* const* d_in, const int* in_sizes, int n_in,
                              void* d_out, int out_size, void* d_ws, size_t ws_size,
                              hipStream_t stream) {
  (void)in_sizes; (void)n_in; (void)out_size; (void)ws_size;
  const float* x       = (const float*)d_in[0];
  const int*   lengths = (const int*)d_in[1];
  const float* Wih0 = (const float*)d_in[2];
  const float* Whh0 = (const float*)d_in[3];
  const float* bih0 = (const float*)d_in[4];
  const float* bhh0 = (const float*)d_in[5];
  const float* g_ih0 = (const float*)d_in[6];
  const float* b_ih0 = (const float*)d_in[7];
  const float* g_hh0 = (const float*)d_in[8];
  const float* b_hh0 = (const float*)d_in[9];
  const float* g_ho0 = (const float*)d_in[10];
  const float* b_ho0 = (const float*)d_in[11];
  const float* Wih1 = (const float*)d_in[12];
  const float* Whh1 = (const float*)d_in[13];
  const float* bih1 = (const float*)d_in[14];
  const float* bhh1 = (const float*)d_in[15];
  const float* g_ih1 = (const float*)d_in[16];
  const float* b_ih1 = (const float*)d_in[17];
  const float* g_hh1 = (const float*)d_in[18];
  const float* b_hh1 = (const float*)d_in[19];
  const float* g_ho1 = (const float*)d_in[20];
  const float* b_ho1 = (const float*)d_in[21];
  const float* fcW = (const float*)d_in[22];
  const float* fcb = (const float*)d_in[23];
  float* out = (float*)d_out;

  // Workspace layout.
  char* ws = (char*)d_ws;
  int* cnt = (int*)ws;
  int* gen = cnt + 1;
  __bf16* Wih0b = (__bf16*)(ws + 256);
  __bf16* Whh0b = Wih0b + (size_t)3 * 512 * 64;
  __bf16* Wih1b = Whh0b + (size_t)3 * 512 * 512;
  __bf16* Whh1b = Wih1b + (size_t)3 * 512 * 512;
  float* u_raw    = (float*)(Whh1b + (size_t)3 * 512 * 512);
  float* w_raw    = u_raw + 64 * 512;
  float* last_raw = w_raw + 64 * 512;

  const int nconv = 3 * 512 * 512;
  conv_weights<<<(nconv + 255) / 256, 256, 0, stream>>>(
      Wih0, Whh0, Wih1, Whh1, Wih0b, Whh0b, Wih1b, Whh1b, cnt, gen);

  gru_persistent<<<32, 128, 0, stream>>>(
      x, lengths,
      bih0, bhh0, g_ih0, b_ih0, g_hh0, b_hh0, g_ho0, b_ho0,
      bih1, bhh1, g_ih1, b_ih1, g_hh1, b_hh1, g_ho1, b_ho1,
      Wih0b, Whh0b, Wih1b, Whh1b,
      u_raw, w_raw, last_raw, cnt, gen);

  fc_final<<<1, 64, 0, stream>>>(last_raw, g_ho1, b_ho1, fcW, fcb, out);
}
